// SwinBlock_45397804319463
// MI455X (gfx1250) — matrix operand order repaired and checked by reading, not compiled, not run on hardware
//
#include <hip/hip_runtime.h>
#include <hip/hip_bf16.h>
#include <math.h>

typedef __bf16 bf16_t;
typedef __attribute__((ext_vector_type(16))) __bf16 v16bf;
typedef __attribute__((ext_vector_type(8)))  float  v8f;
typedef __attribute__((ext_vector_type(4)))  unsigned int u32x4;
typedef __attribute__((ext_vector_type(4)))  int v4i;

#define AS1 __attribute__((address_space(1)))
#define AS3 __attribute__((address_space(3)))

#if defined(__gfx1250__) && __has_builtin(__builtin_amdgcn_global_load_async_to_lds_b128) && \
    __has_builtin(__builtin_amdgcn_s_wait_asynccnt)
#define USE_ASYNC_LDS 1
#else
#define USE_ASYNC_LDS 0
#endif

union FragU { v16bf v; u32x4 q[2]; };

static __device__ __forceinline__ unsigned short f2bf_bits(float f) {
    unsigned u = __float_as_uint(f);
    unsigned r = u + 0x7FFFu + ((u >> 16) & 1u);   // round-to-nearest-even
    return (unsigned short)(r >> 16);
}
static __device__ __forceinline__ bf16_t f2bf(float f) {
    unsigned short s = f2bf_bits(f);
    bf16_t r; __builtin_memcpy(&r, &s, 2); return r;
}
static __device__ __forceinline__ unsigned pk_bf2(float a, float b) {
    return (unsigned)f2bf_bits(a) | ((unsigned)f2bf_bits(b) << 16);
}

static __device__ __forceinline__ v8f wmma_bf16(v16bf a, v16bf b, v8f c) {
    return __builtin_amdgcn_wmma_f32_16x16x32_bf16(false, a, false, b, (short)0, c, false, false);
}

// A fragment 16x32 bf16 from row-major [rows x ld] matrix.
// Lane layout (ISA 7.12.2): lanes 0-15 row M=l, K = kb..kb+7 & kb+16..kb+23 with kb=kTile*32
// lanes 16-31: same rows, kb += 8.
static __device__ __forceinline__ v16bf load_a_frag(const bf16_t* base, int row, int ld,
                                                    int ktile_base, int lane) {
    int kb = ktile_base + ((lane & 16) ? 8 : 0);
    const bf16_t* p = base + (size_t)row * ld + kb;
    FragU f;
    f.q[0] = *(const u32x4*)(p);
    f.q[1] = *(const u32x4*)(p + 16);
    return f.v;
}
// B fragment 32x16 bf16 from K-transposed weights Wt[N][K] (row-major, contiguous in K).
// lanes 0-15: col N=l, K = kb..kb+7 & kb+16..kb+23 ; lanes 16-31: col N=l-16, kb += 8.
static __device__ __forceinline__ v16bf load_b_frag(const bf16_t* base, int col, int ld,
                                                    int ktile_base, int lane) {
    int kb = ktile_base + ((lane & 16) ? 8 : 0);
    const bf16_t* p = base + (size_t)col * ld + kb;
    FragU f;
    f.q[0] = *(const u32x4*)(p);
    f.q[1] = *(const u32x4*)(p + 16);
    return f.v;
}

// Stage 64x256 bf16 (32 KB) from global into LDS. Prefer the gfx1250 direct
// async global->LDS path (ASYNCcnt) so the copy bypasses VGPRs.
static __device__ __forceinline__ void stage_tile_to_lds(const bf16_t* __restrict__ src,
                                                         bf16_t* __restrict__ dst, int tid) {
#if USE_ASYNC_LDS
    for (int i = tid; i < 2048; i += 256) {
        __builtin_amdgcn_global_load_async_to_lds_b128(
            (AS1 v4i*)(src + (size_t)i * 8),
            (AS3 v4i*)(dst + (size_t)i * 8), 0, 0);
    }
    __builtin_amdgcn_s_wait_asynccnt(0);
#else
    const u32x4* s = (const u32x4*)src;
    u32x4* d = (u32x4*)dst;
    for (int i = tid; i < 2048; i += 256) d[i] = s[i];
#endif
}

// ---------------------------------------------------------------------------
// Kernel 1: f32 -> bf16 with transpose:  in[K][N] -> out[N][K]
__global__ __launch_bounds__(256)
void k_w2bf_t(const float* __restrict__ in, bf16_t* __restrict__ out, int K, int N) {
    int idx = blockIdx.x * 256 + threadIdx.x;
    if (idx >= K * N) return;
    int k = idx / N, n = idx - k * N;
    out[(size_t)n * K + k] = f2bf(in[idx]);
}

// ---------------------------------------------------------------------------
// Kernel 2: AdaLN scale/shift:  ss[b][0:512] = t@a1w + a1b ; ss[b][512:1024] = t@a2w + a2b
__global__ __launch_bounds__(256)
void k_adaln(const float* __restrict__ t, const float* __restrict__ w1,
             const float* __restrict__ b1, const float* __restrict__ w2,
             const float* __restrict__ b2, float* __restrict__ ss) {
    int idx = blockIdx.x * 256 + threadIdx.x;   // 8 * 1024
    int b = idx >> 10, j = idx & 1023;
    const float* w = (j < 512) ? w1 : w2;
    const float* bb = (j < 512) ? b1 : b2;
    int jj = j & 511;
    float acc = bb[jj];
    const float* tr = t + b * 256;
#pragma unroll 4
    for (int r = 0; r < 256; ++r) acc += tr[r] * w[r * 512 + jj];
    ss[idx] = acc;
}

// ---------------------------------------------------------------------------
// Kernel 3/5: LayerNorm + AdaLN modulate, optional shifted-window permutation, f32 -> bf16.
// One wave per token row (32 lanes x 8 channels). shifted=1: output row is window-token order.
__global__ __launch_bounds__(256)
void k_ln_mod(const float* __restrict__ xin, const float* __restrict__ ss,
              bf16_t* __restrict__ xout, int shifted) {
    const int row = blockIdx.x * 8 + (threadIdx.x >> 5);
    const int l = threadIdx.x & 31;
    const int b = row >> 12;                      // 4096 rows per batch
    int row_in = row;
    if (shifted) {
        int rem = row & 4095;
        int wloc = rem >> 6, tk = rem & 63;
        int wy = wloc >> 3, wx = wloc & 7, iy = tk >> 3, ix = tk & 7;
        int y = (wy * 8 + iy + 4) & 63;
        int x = (wx * 8 + ix + 4) & 63;
        row_in = (b << 12) + y * 64 + x;
    }
    const float4* src = (const float4*)(xin + (size_t)row_in * 256) + l * 2;
    float4 a0 = src[0], a1 = src[1];
    float s  = a0.x + a0.y + a0.z + a0.w + a1.x + a1.y + a1.z + a1.w;
    float s2 = a0.x*a0.x + a0.y*a0.y + a0.z*a0.z + a0.w*a0.w
             + a1.x*a1.x + a1.y*a1.y + a1.z*a1.z + a1.w*a1.w;
#pragma unroll
    for (int m = 16; m > 0; m >>= 1) {
        s  += __shfl_xor(s,  m, 32);
        s2 += __shfl_xor(s2, m, 32);
    }
    float mu  = s  * (1.f / 256.f);
    float var = s2 * (1.f / 256.f) - mu * mu;
    float rs  = rsqrtf(var + 1e-5f);
    const float* mod = ss + b * 1024 + (shifted ? 0 : 512);   // scale[ch], shift at +256
    int ch0 = l * 8;
    float o[8] = {a0.x, a0.y, a0.z, a0.w, a1.x, a1.y, a1.z, a1.w};
    unsigned pk[4];
#pragma unroll
    for (int e = 0; e < 4; ++e) {
        int c0 = ch0 + 2 * e;
        float r0 = (o[2*e]   - mu) * rs * (1.f + mod[c0])     + mod[256 + c0];
        float r1 = (o[2*e+1] - mu) * rs * (1.f + mod[c0 + 1]) + mod[256 + c0 + 1];
        pk[e] = pk_bf2(r0, r1);
    }
    u32x4 v; v.x = pk[0]; v.y = pk[1]; v.z = pk[2]; v.w = pk[3];
    *(u32x4*)(xout + (size_t)row * 256 + ch0) = v;
}

// ---------------------------------------------------------------------------
// Kernel 4: fused per-window attention. 1 block (8 waves) per window.
// LDS: xw 32K | q 32K | k 32K | vT 32K | ao 32K | logits 16K | attn 8K  = 188416 B
__global__ __launch_bounds__(256)
void k_attn(const bf16_t* __restrict__ xw,
            const bf16_t* __restrict__ wqt, const bf16_t* __restrict__ wkt,
            const bf16_t* __restrict__ wvt, const bf16_t* __restrict__ wot,
            const float* __restrict__ qbias, const float* __restrict__ kbias,
            const float* __restrict__ vbias, const float* __restrict__ obias,
            const float* __restrict__ btab,
            const float* __restrict__ x_in, float* __restrict__ x1) {
    extern __shared__ char smem[];
    bf16_t* xw_s = (bf16_t*)smem;              // [64][256]
    bf16_t* q_s  = xw_s + 64 * 256;            // [64][256]
    bf16_t* k_s  = q_s  + 64 * 256;            // [64][256]
    bf16_t* vT_s = k_s  + 64 * 256;            // [256][64]  (channel-major)
    bf16_t* ao_s = vT_s + 64 * 256;            // [64][256]
    float*  lg_s = (float*)(ao_s + 64 * 256);  // [64][64]
    bf16_t* at_s = (bf16_t*)(lg_s + 64 * 64);  // [64][64]

    const int tid = threadIdx.x, wv = tid >> 5, l = tid & 31, l15 = l & 15;
    const int hi8 = (l & 16) ? 8 : 0;
    const int w = blockIdx.x;

    // stage window activations into LDS (async direct-to-LDS path)
    stage_tile_to_lds(xw + (size_t)w * 64 * 256, xw_s, tid);
    __syncthreads();

    // ---- QKV projections: 3 mats x (4x16) tiles = 192 jobs over 8 waves ----
    for (int j = wv; j < 192; j += 8) {
        int mat = j / 64, t = j % 64, mT = t >> 4, nT = t & 15;
        const bf16_t* wt = (mat == 0) ? wqt : (mat == 1) ? wkt : wvt;
        const float* bia = (mat == 0) ? qbias : (mat == 1) ? kbias : vbias;
        int n = nT * 16 + l15;
        float bv = bia[n];
        v8f c;
#pragma unroll
        for (int r = 0; r < 8; ++r) c[r] = bv;
        int row = mT * 16 + l15;
#pragma unroll
        for (int k8 = 0; k8 < 8; ++k8) {
            if (k8 < 7) __builtin_prefetch(wt + (size_t)n * 256 + (k8 + 1) * 32, 0, 1);
            v16bf a = load_a_frag(xw_s, row, 256, k8 * 32, l);
            v16bf bf = load_b_frag(wt, n, 256, k8 * 32, l);
            c = wmma_bf16(a, bf, c);
        }
        int tok = mT * 16 + hi8;
        if (mat == 2) { // V stored transposed: vT[n][tok..tok+7] -> one 16B ds store
            u32x4 pk;
            pk.x = pk_bf2(c[0], c[1]); pk.y = pk_bf2(c[2], c[3]);
            pk.z = pk_bf2(c[4], c[5]); pk.w = pk_bf2(c[6], c[7]);
            *(u32x4*)(vT_s + n * 64 + tok) = pk;
        } else {
            bf16_t* dst = (mat == 0) ? q_s : k_s;
#pragma unroll
            for (int r = 0; r < 8; ++r) dst[(tok + r) * 256 + n] = f2bf(c[r]);
        }
    }
    __syncthreads();

    const float scl = 0.17677669529663687f;    // 1/sqrt(32)

    // ---- per-head attention ----
    for (int h = 0; h < 8; ++h) {
        // logits = Q Kt : 16 tiles, K-dim=32 -> single WMMA each
        for (int j = wv; j < 16; j += 8) {
            int mT = j >> 2, nT = j & 3;
            int row = mT * 16 + l15;
            int n   = nT * 16 + l15;
            v16bf a = load_a_frag(q_s, row, 256, h * 32, l);
            v16bf bf = load_b_frag(k_s, n, 256, h * 32, l);
            v8f c;
#pragma unroll
            for (int r = 0; r < 8; ++r) c[r] = 0.f;
            c = wmma_bf16(a, bf, c);
            int tok = mT * 16 + hi8;
#pragma unroll
            for (int r = 0; r < 8; ++r) lg_s[(tok + r) * 64 + n] = c[r];
        }
        __syncthreads();
        // softmax + relative-position bias, rows wv*8..wv*8+7, 2 cols/lane
        for (int rr = 0; rr < 8; ++rr) {
            int qi = wv * 8 + rr;
            int qy = qi >> 3, qx = qi & 7;
            int c0 = l, c1 = l + 32;
            int ky0 = c0 >> 3, kx0 = c0 & 7, ky1 = c1 >> 3, kx1 = c1 & 7;
            float v0 = lg_s[qi * 64 + c0] * scl
                     + btab[((qy - ky0 + 7) * 15 + (qx - kx0 + 7)) * 8 + h];
            float v1 = lg_s[qi * 64 + c1] * scl
                     + btab[((qy - ky1 + 7) * 15 + (qx - kx1 + 7)) * 8 + h];
            float mx = fmaxf(v0, v1);
#pragma unroll
            for (int m = 16; m > 0; m >>= 1) mx = fmaxf(mx, __shfl_xor(mx, m, 32));
            float e0 = __expf(v0 - mx), e1 = __expf(v1 - mx);
            float sm = e0 + e1;
#pragma unroll
            for (int m = 16; m > 0; m >>= 1) sm += __shfl_xor(sm, m, 32);
            float inv = 1.0f / sm;
            at_s[qi * 64 + c0] = f2bf(e0 * inv);
            at_s[qi * 64 + c1] = f2bf(e1 * inv);
        }
        __syncthreads();
        // attn @ V : 4m x 2n tiles, K=64 -> 2 WMMA; one tile per wave
        for (int j = wv; j < 8; j += 8) {
            int mT = j >> 1, nT = j & 1;
            int row = mT * 16 + l15;
            int d = nT * 16 + l15;
            v8f c;
#pragma unroll
            for (int r = 0; r < 8; ++r) c[r] = 0.f;
#pragma unroll
            for (int k8 = 0; k8 < 2; ++k8) {
                v16bf a = load_a_frag(at_s, row, 64, k8 * 32, l);
                v16bf bf = load_b_frag(vT_s, h * 32 + d, 64, k8 * 32, l);
                c = wmma_bf16(a, bf, c);
            }
            int tok = mT * 16 + hi8;
#pragma unroll
            for (int r = 0; r < 8; ++r) ao_s[(tok + r) * 256 + h * 32 + d] = f2bf(c[r]);
        }
        __syncthreads();
    }

    // ---- O-projection + window reverse + residual ----
    int b = w >> 6, wloc = w & 63, wy = wloc >> 3, wx = wloc & 7;
    for (int j = wv; j < 64; j += 8) {
        int mT = j >> 4, nT = j & 15;
        int n = nT * 16 + l15;
        float bv = obias[n];
        v8f c;
#pragma unroll
        for (int r = 0; r < 8; ++r) c[r] = bv;
        int row = mT * 16 + l15;
#pragma unroll
        for (int k8 = 0; k8 < 8; ++k8) {
            if (k8 < 7) __builtin_prefetch(wot + (size_t)n * 256 + (k8 + 1) * 32, 0, 1);
            v16bf a = load_a_frag(ao_s, row, 256, k8 * 32, l);
            v16bf bf = load_b_frag(wot, n, 256, k8 * 32, l);
            c = wmma_bf16(a, bf, c);
        }
        int tokb = mT * 16 + hi8;
#pragma unroll
        for (int r = 0; r < 8; ++r) {
            int tok = tokb + r;
            int iy = tok >> 3, ix = tok & 7;
            int y = (wy * 8 + iy + 4) & 63;
            int x = (wx * 8 + ix + 4) & 63;
            size_t gi = (((size_t)b << 12) + y * 64 + x) * 256 + n;
            x1[gi] = x_in[gi] + c[r];
        }
    }
}

// ---------------------------------------------------------------------------
// Kernel 6: fused FFN: out = x1 + GELU(xn2@W1+b1)@W2+b2. 64 rows per block,
// hidden computed in 4 chunks of 256 staged through LDS, out C-frags persistent.
__global__ __launch_bounds__(256)
void k_ffn(const bf16_t* __restrict__ xn2, const bf16_t* __restrict__ w1t,
           const bf16_t* __restrict__ w2t, const float* __restrict__ b1,
           const float* __restrict__ b2, const float* __restrict__ x1,
           float* __restrict__ out) {
    extern __shared__ char smem[];
    bf16_t* xs = (bf16_t*)smem;      // [64][256] input tile
    bf16_t* hs = xs + 64 * 256;      // [64][256] hidden chunk
    const int tid = threadIdx.x, wv = tid >> 5, l = tid & 31, l15 = l & 15;
    const int hi8 = (l & 16) ? 8 : 0;
    const size_t m0 = (size_t)blockIdx.x * 64;

    stage_tile_to_lds(xn2 + m0 * 256, xs, tid);
    __syncthreads();

    v8f cacc[8];
    int tmT[8], tnT[8];
#pragma unroll
    for (int i = 0; i < 8; ++i) {
        int j = wv + 8 * i;
        tmT[i] = j >> 4; tnT[i] = j & 15;
        float bv = b2[tnT[i] * 16 + l15];
#pragma unroll
        for (int r = 0; r < 8; ++r) cacc[i][r] = bv;
    }

    for (int cc = 0; cc < 4; ++cc) {
        // produce GELU(hidden) chunk into LDS
        for (int i = 0; i < 8; ++i) {
            int j = wv + 8 * i;
            int mT = j >> 4, nT = j & 15;
            int hcol = cc * 256 + nT * 16 + l15;
            float bv = b1[hcol];
            v8f c;
#pragma unroll
            for (int r = 0; r < 8; ++r) c[r] = bv;
            int row = mT * 16 + l15;
#pragma unroll
            for (int k8 = 0; k8 < 8; ++k8) {
                if (k8 < 7) __builtin_prefetch(w1t + (size_t)hcol * 256 + (k8 + 1) * 32, 0, 1);
                v16bf a = load_a_frag(xs, row, 256, k8 * 32, l);
                v16bf bf = load_b_frag(w1t, hcol, 256, k8 * 32, l);
                c = wmma_bf16(a, bf, c);
            }
            int tok = mT * 16 + hi8;
            int ncl = nT * 16 + l15;
#pragma unroll
            for (int r = 0; r < 8; ++r) {
                float v = c[r];
                float g = 0.5f * v * (1.f + erff(v * 0.70710678118654752f));
                hs[(tok + r) * 256 + ncl] = f2bf(g);
            }
        }
        __syncthreads();
        // consume: accumulate out tiles
        for (int i = 0; i < 8; ++i) {
            int n = tnT[i] * 16 + l15;
            int row = tmT[i] * 16 + l15;
#pragma unroll
            for (int k8 = 0; k8 < 8; ++k8) {
                v16bf a = load_a_frag(hs, row, 256, k8 * 32, l);
                v16bf bf = load_b_frag(w2t, n, 1024, cc * 256 + k8 * 32, l);
                cacc[i] = wmma_bf16(a, bf, cacc[i]);
            }
        }
        __syncthreads();
    }

#pragma unroll
    for (int i = 0; i < 8; ++i) {
        int n = tnT[i] * 16 + l15;
        int tokb = tmT[i] * 16 + hi8;
#pragma unroll
        for (int r = 0; r < 8; ++r) {
            size_t gi = (m0 + tokb + r) * 256 + n;
            out[gi] = cacc[i][r] + x1[gi];
        }
    }
}

// ---------------------------------------------------------------------------
extern "C" void kernel_launch(void* const* d_in, const int* in_sizes, int n_in,
                              void* d_out, int out_size, void* d_ws, size_t ws_size,
                              hipStream_t stream) {
    const float* x    = (const float*)d_in[0];
    const float* temb = (const float*)d_in[1];
    // d_in[2]=H, d_in[3]=W (fixed 64)
    const float* a1w = (const float*)d_in[4];
    const float* a1b = (const float*)d_in[5];
    const float* a2w = (const float*)d_in[6];
    const float* a2b = (const float*)d_in[7];
    const float* btab = (const float*)d_in[8];
    const float* qw = (const float*)d_in[9];   const float* qb = (const float*)d_in[10];
    const float* kw = (const float*)d_in[11];  const float* kb = (const float*)d_in[12];
    const float* vw = (const float*)d_in[13];  const float* vb = (const float*)d_in[14];
    const float* ow = (const float*)d_in[15];  const float* ob = (const float*)d_in[16];
    const float* f1w = (const float*)d_in[17]; const float* f1b = (const float*)d_in[18];
    const float* f2w = (const float*)d_in[19]; const float* f2b = (const float*)d_in[20];

    char* wsb = (char*)d_ws;
    size_t off = 0;
    float*  ss  = (float*)(wsb + off);  off += 8 * 1024 * 4;        // 32 KB
    bf16_t* wqt = (bf16_t*)(wsb + off); off += 65536 * 2;
    bf16_t* wkt = (bf16_t*)(wsb + off); off += 65536 * 2;
    bf16_t* wvt = (bf16_t*)(wsb + off); off += 65536 * 2;
    bf16_t* wot = (bf16_t*)(wsb + off); off += 65536 * 2;
    bf16_t* w1t = (bf16_t*)(wsb + off); off += 262144 * 2;
    bf16_t* w2t = (bf16_t*)(wsb + off); off += 262144 * 2;
    bf16_t* xw  = (bf16_t*)(wsb + off); off += (size_t)32768 * 256 * 2;  // 16 MB
    float*  x1  = (float*)(wsb + off);  off += (size_t)32768 * 256 * 4;  // 32 MB
    bf16_t* xn2 = (bf16_t*)(wsb + off); off += (size_t)32768 * 256 * 2;  // 16 MB

    // 1. weight conversion (f32 -> bf16, K-transposed for contiguous B-fragments)
    k_w2bf_t<<<(256 * 256) / 256, 256, 0, stream>>>(qw, wqt, 256, 256);
    k_w2bf_t<<<(256 * 256) / 256, 256, 0, stream>>>(kw, wkt, 256, 256);
    k_w2bf_t<<<(256 * 256) / 256, 256, 0, stream>>>(vw, wvt, 256, 256);
    k_w2bf_t<<<(256 * 256) / 256, 256, 0, stream>>>(ow, wot, 256, 256);
    k_w2bf_t<<<(256 * 1024) / 256, 256, 0, stream>>>(f1w, w1t, 256, 1024);
    k_w2bf_t<<<(1024 * 256) / 256, 256, 0, stream>>>(f2w, w2t, 1024, 256);

    // 2. AdaLN scale/shift
    k_adaln<<<32, 256, 0, stream>>>(temb, a1w, a1b, a2w, a2b, ss);

    // 3. LN1 + modulate + shifted-window partition
    k_ln_mod<<<4096, 256, 0, stream>>>(x, ss, xw, 1);

    // 4. fused window attention (+O-proj, +residual, +window reverse)
    size_t attn_lds = 5 * 64 * 256 * 2 + 64 * 64 * 4 + 64 * 64 * 2;  // 188416 B
    k_attn<<<512, 256, attn_lds, stream>>>(xw, wqt, wkt, wvt, wot,
                                           qb, kb, vb, ob, btab, x, x1);

    // 5. LN2 + modulate
    k_ln_mod<<<4096, 256, 0, stream>>>(x1, ss, xn2, 0);

    // 6. fused FFN (+residual) -> d_out
    size_t ffn_lds = 2 * 64 * 256 * 2;  // 64 KB
    k_ffn<<<512, 256, ffn_lds, stream>>>(xn2, w1t, w2t, f1b, f2b, x1, (float*)d_out);
}
